// MFI_15015205667036
// MI455X (gfx1250) — matrix operand order repaired and checked by reading, not compiled, not run on hardware
//
#include <hip/hip_runtime.h>
#include <hip/hip_bf16.h>

typedef __attribute__((ext_vector_type(16))) _Float16 v16h;
typedef __attribute__((ext_vector_type(8)))  _Float16 v8h;
typedef __attribute__((ext_vector_type(8)))  float    v8f;

#define NF 3
#define BB 4
#define CC 64
#define DD 8
#define NN 4096  // H*W

// ---------------------------------------------------------------------------
// Projection: q = wq@x+bq, k = wk@x+bk (f16 [B,N,8]); v = wv@xv+bv (f16 [B,C,N])
// ---------------------------------------------------------------------------
__global__ void __launch_bounds__(256)
proj_qkv_kernel(const float* __restrict__ xq, const float* __restrict__ xv,
                const float* __restrict__ wq, const float* __restrict__ bq,
                const float* __restrict__ wk, const float* __restrict__ bk,
                const float* __restrict__ wv, const float* __restrict__ bv,
                _Float16* __restrict__ qf, _Float16* __restrict__ kf,
                _Float16* __restrict__ vf)
{
    int t = blockIdx.x * blockDim.x + threadIdx.x;   // 0 .. B*N-1
    int b = t / NN, n = t % NN;
    const float* xqb = xq + (size_t)b * CC * NN + n;
    const float* xvb = xv + (size_t)b * CC * NN + n;

    float qa[DD], ka[DD];
#pragma unroll
    for (int d = 0; d < DD; ++d) { qa[d] = bq[d]; ka[d] = bk[d]; }
    for (int c = 0; c < CC; ++c) {
        float xc = xqb[(size_t)c * NN];
#pragma unroll
        for (int d = 0; d < DD; ++d) {
            qa[d] = fmaf(wq[d * CC + c], xc, qa[d]);
            ka[d] = fmaf(wk[d * CC + c], xc, ka[d]);
        }
    }
    _Float16* qrow = qf + (size_t)(b * NN + n) * DD;
    _Float16* krow = kf + (size_t)(b * NN + n) * DD;
#pragma unroll
    for (int d = 0; d < DD; ++d) { qrow[d] = (_Float16)qa[d]; krow[d] = (_Float16)ka[d]; }

    // v projection in 4 chunks of 16 output channels (re-stream xv; L2 resident)
    for (int e0 = 0; e0 < CC; e0 += 16) {
        float acc[16];
#pragma unroll
        for (int i = 0; i < 16; ++i) acc[i] = bv[e0 + i];
        for (int c = 0; c < CC; ++c) {
            float xc = xvb[(size_t)c * NN];
#pragma unroll
            for (int i = 0; i < 16; ++i)
                acc[i] = fmaf(wv[(e0 + i) * CC + c], xc, acc[i]);
        }
#pragma unroll
        for (int i = 0; i < 16; ++i)
            vf[(size_t)b * CC * NN + (size_t)(e0 + i) * NN + n] = (_Float16)acc[i];
    }
}

// ---------------------------------------------------------------------------
// Flash attention (wave32 WMMA). Each wave owns 16 query rows, streams m in
// chunks of 32.  S^T = K Q^T is computed (A = K-tile, B = Q^T) so that each
// lane's C-fragment holds 8 m-values of ONE n-column: softmax reductions are
// in-register trees + a single shfl_xor(16), not per-row lane reductions.
// ---------------------------------------------------------------------------
__global__ void __launch_bounds__(256)
flash_attn_kernel(const _Float16* __restrict__ qf,   // [B,N,8]
                  const _Float16* __restrict__ kf,   // [B,N,8]
                  const _Float16* __restrict__ vf,   // [B,C,N]
                  const float* __restrict__ resid,   // [B,C,N]
                  float* __restrict__ out)           // [B,C,N]
{
    __shared__ __align__(16) _Float16 psh[8][16][32];   // per-wave P tile [n][m]
    const int wave = threadIdx.x >> 5;
    const int lane = threadIdx.x & 31;
    const int lcol = lane & 15;   // column / row-within-16 index
    const int hs   = lane >> 4;   // half-wave select

    const int tiles_per_b = NN / 128;                  // 128 q-rows per block
    const int b  = blockIdx.x / tiles_per_b;
    const int n0 = (blockIdx.x % tiles_per_b) * 128 + wave * 16;

    // ---- Q^T B-fragment: column n = lcol, K(d) = comps + 16*hs (seq layout) ----
    v16h bq;
#pragma unroll
    for (int j = 0; j < 16; ++j) bq[j] = (_Float16)0.0f;
    if (hs == 0) {  // d = 0..7 real, 8..31 zero padding
        v8h qrow = *(const v8h*)(qf + (size_t)(b * NN + n0 + lcol) * DD);
#pragma unroll
        for (int j = 0; j < 8; ++j) bq[j] = qrow[j];
    }

    v8f O0 = {}, O1 = {}, O2 = {}, O3 = {};
    float m_i = -3.0e38f, l_i = 0.0f;     // running stats for column n = lcol

    const _Float16* vbase = vf + (size_t)b * CC * NN;

    for (int m0 = 0; m0 < NN; m0 += 32) {
        // ---- S^T = K Q^T, two 16-m-row tiles (A = K rows, d padded to 32) ----
        v16h ak0, ak1;
#pragma unroll
        for (int j = 0; j < 16; ++j) { ak0[j] = (_Float16)0.0f; ak1[j] = (_Float16)0.0f; }
        if (hs == 0) {  // A layout: lanes 0-15 comps 0..7 = K(d)=0..7
            v8h kr0 = *(const v8h*)(kf + (size_t)(b * NN + m0 + lcol) * DD);
            v8h kr1 = *(const v8h*)(kf + (size_t)(b * NN + m0 + 16 + lcol) * DD);
#pragma unroll
            for (int j = 0; j < 8; ++j) { ak0[j] = kr0[j]; ak1[j] = kr1[j]; }
        }
        v8f ST0 = {}, ST1 = {};  // lane: ST0[r] = S[n=lcol][m = m0 + r+8hs]
        ST0 = __builtin_amdgcn_wmma_f32_16x16x32_f16(false, ak0, false, bq, (short)0, ST0, false, false);
        ST1 = __builtin_amdgcn_wmma_f32_16x16x32_f16(false, ak1, false, bq, (short)0, ST1, false, false);

        // ---- online softmax for column n = lcol (in-lane tree + 1 shuffle) ----
        float cm = fmaxf(ST0[0], ST1[0]);
#pragma unroll
        for (int r = 1; r < 8; ++r) cm = fmaxf(cm, fmaxf(ST0[r], ST1[r]));
        cm = fmaxf(cm, __shfl_xor(cm, 16, 32));
        float mn    = fmaxf(m_i, cm);
        float scale = __expf(m_i - mn);
        m_i = mn;

        float p0[8], p1[8], rs = 0.0f;
#pragma unroll
        for (int r = 0; r < 8; ++r) {
            p0[r] = __expf(ST0[r] - mn);
            p1[r] = __expf(ST1[r] - mn);
            rs += p0[r] + p1[r];
        }
        rs += __shfl_xor(rs, 16, 32);
        l_i = l_i * scale + rs;

        // ---- rescale O: broadcast per-row scale (rows n = r + 8*hs) ----
#pragma unroll
        for (int r = 0; r < 8; ++r) {
            float sr = __shfl(scale, r + 8 * hs, 32);
            O0[r] *= sr; O1[r] *= sr; O2[r] *= sr; O3[r] *= sr;
        }

        // ---- P^T fragment -> LDS [n][m] as two packed b128 stores ----
        v8h pk0, pk1;
#pragma unroll
        for (int r = 0; r < 8; ++r) { pk0[r] = (_Float16)p0[r]; pk1[r] = (_Float16)p1[r]; }
        *(v8h*)(&psh[wave][lcol][8 * hs])      = pk0;   // m_local 8hs .. 8hs+7
        *(v8h*)(&psh[wave][lcol][16 + 8 * hs]) = pk1;   // m_local 16+8hs ..
        __syncthreads();

        // ---- read P as A-fragment (contiguous K: m_local = 16*hs + j) ----
        const _Float16* prow = &psh[wave][lcol][0];
        v8h plo = *(const v8h*)(prow + (hs ? 16 : 0));
        v8h phi = *(const v8h*)(prow + (hs ? 24 : 8));
        v16h ap;
#pragma unroll
        for (int j = 0; j < 8; ++j) { ap[j] = plo[j]; ap[8 + j] = phi[j]; }
        __syncthreads();

        // ---- O += P V : 4 e-tiles; B-fragment = 16 contiguous m per lane ----
        const size_t moff = (size_t)m0 + 16 * hs;
        v16h bv0 = *(const v16h*)(vbase + (size_t)(lcol)      * NN + moff);
        v16h bv1 = *(const v16h*)(vbase + (size_t)(16 + lcol) * NN + moff);
        v16h bv2 = *(const v16h*)(vbase + (size_t)(32 + lcol) * NN + moff);
        v16h bv3 = *(const v16h*)(vbase + (size_t)(48 + lcol) * NN + moff);
        O0 = __builtin_amdgcn_wmma_f32_16x16x32_f16(false, ap, false, bv0, (short)0, O0, false, false);
        O1 = __builtin_amdgcn_wmma_f32_16x16x32_f16(false, ap, false, bv1, (short)0, O1, false, false);
        O2 = __builtin_amdgcn_wmma_f32_16x16x32_f16(false, ap, false, bv2, (short)0, O2, false, false);
        O3 = __builtin_amdgcn_wmma_f32_16x16x32_f16(false, ap, false, bv3, (short)0, O3, false, false);
    }

    // ---- normalize (broadcast 1/l per row), residual, store fp32 [B,C,N] ----
#pragma unroll
    for (int r = 0; r < 8; ++r) {
        float lr  = __shfl(l_i, r + 8 * hs, 32);
        float inv = 1.0f / lr;
        size_t base = (size_t)b * CC * NN + (size_t)(n0 + r + 8 * hs);
        size_t i0 = base + (size_t)(lcol)      * NN;
        size_t i1 = base + (size_t)(16 + lcol) * NN;
        size_t i2 = base + (size_t)(32 + lcol) * NN;
        size_t i3 = base + (size_t)(48 + lcol) * NN;
        out[i0] = O0[r] * inv + resid[i0];
        out[i1] = O1[r] * inv + resid[i1];
        out[i2] = O2[r] * inv + resid[i2];
        out[i3] = O3[r] * inv + resid[i3];
    }
}

// ---------------------------------------------------------------------------
// Fusion: out = (tfus_w @ [mfi0;mfi1] + tfus_b) * sa1 + sa1
// ---------------------------------------------------------------------------
__global__ void __launch_bounds__(256)
fusion_kernel(const float* __restrict__ mfi0, const float* __restrict__ mfi1,
              const float* __restrict__ w,    // [64][128]
              const float* __restrict__ bias, // [64]
              const float* __restrict__ sa1,  // [B,C,N]
              float* __restrict__ out)        // [B,C,N]
{
    int t = blockIdx.x * blockDim.x + threadIdx.x;
    int b = t / NN, n = t % NN;
    float acc[CC];
#pragma unroll
    for (int o = 0; o < CC; ++o) acc[o] = bias[o];
    const float* m0b = mfi0 + (size_t)b * CC * NN + n;
    const float* m1b = mfi1 + (size_t)b * CC * NN + n;
    for (int c = 0; c < CC; ++c) {
        float x0 = m0b[(size_t)c * NN];
        float x1 = m1b[(size_t)c * NN];
#pragma unroll
        for (int o = 0; o < CC; ++o) {
            acc[o] = fmaf(w[o * 2 * CC + c], x0, acc[o]);
            acc[o] = fmaf(w[o * 2 * CC + CC + c], x1, acc[o]);
        }
    }
    size_t base = (size_t)b * CC * NN + n;
    for (int o = 0; o < CC; ++o) {
        float s = sa1[base + (size_t)o * NN];
        out[base + (size_t)o * NN] = fmaf(acc[o], s, s);
    }
}

// ---------------------------------------------------------------------------
extern "C" void kernel_launch(void* const* d_in, const int* in_sizes, int n_in,
                              void* d_out, int out_size, void* d_ws, size_t ws_size,
                              hipStream_t stream)
{
    const float* x      = (const float*)d_in[0];
    const float* sa_wq  = (const float*)d_in[1];
    const float* sa_bq  = (const float*)d_in[2];
    const float* sa_wk  = (const float*)d_in[3];
    const float* sa_bk  = (const float*)d_in[4];
    const float* sa_wv  = (const float*)d_in[5];
    const float* sa_bv  = (const float*)d_in[6];
    const float* ga_wq  = (const float*)d_in[7];
    const float* ga_bq  = (const float*)d_in[8];
    const float* ga_wk  = (const float*)d_in[9];
    const float* ga_bk  = (const float*)d_in[10];
    const float* ga_wv  = (const float*)d_in[11];
    const float* ga_bv  = (const float*)d_in[12];
    const float* tfus_w = (const float*)d_in[13];
    const float* tfus_b = (const float*)d_in[14];

    const size_t BCN = (size_t)BB * CC * NN;   // 1,048,576 elements
    char* ws = (char*)d_ws;
    float* sa0  = (float*)(ws);
    float* sa1  = (float*)(ws + 1 * BCN * sizeof(float));
    float* sa2  = (float*)(ws + 2 * BCN * sizeof(float));
    float* mfi0 = (float*)(ws + 3 * BCN * sizeof(float));
    float* mfi1 = (float*)(ws + 4 * BCN * sizeof(float));
    char*  h    = ws + 5 * BCN * sizeof(float);
    _Float16* qf = (_Float16*)(h);
    _Float16* kf = (_Float16*)(h + (size_t)BB * NN * DD * sizeof(_Float16));
    _Float16* vf = (_Float16*)(h + 2 * (size_t)BB * NN * DD * sizeof(_Float16));

    const int projGrid = (BB * NN) / 256;   // 64
    const int attnGrid = (BB * NN) / 128;   // 128 (8 waves x 16 rows per block)

    float* sa[3] = { sa0, sa1, sa2 };
    for (int f = 0; f < NF; ++f) {
        const float* xf = x + (size_t)f * BCN;
        proj_qkv_kernel<<<projGrid, 256, 0, stream>>>(
            xf, xf,
            sa_wq + f * DD * CC, sa_bq + f * DD,
            sa_wk + f * DD * CC, sa_bk + f * DD,
            sa_wv + f * CC * CC, sa_bv + f * CC,
            qf, kf, vf);
        flash_attn_kernel<<<attnGrid, 256, 0, stream>>>(qf, kf, vf, xf, sa[f]);
    }

    // GA0: q,k from sa0; v + residual from sa1
    proj_qkv_kernel<<<projGrid, 256, 0, stream>>>(
        sa0, sa1, ga_wq, ga_bq, ga_wk, ga_bk, ga_wv, ga_bv, qf, kf, vf);
    flash_attn_kernel<<<attnGrid, 256, 0, stream>>>(qf, kf, vf, sa1, mfi0);

    // GA1: q,k from sa2; v + residual from sa1
    proj_qkv_kernel<<<projGrid, 256, 0, stream>>>(
        sa2, sa1,
        ga_wq + DD * CC, ga_bq + DD,
        ga_wk + DD * CC, ga_bk + DD,
        ga_wv + CC * CC, ga_bv + CC,
        qf, kf, vf);
    flash_attn_kernel<<<attnGrid, 256, 0, stream>>>(qf, kf, vf, sa1, mfi1);

    fusion_kernel<<<projGrid, 256, 0, stream>>>(mfi0, mfi1, tfus_w, tfus_b, sa1,
                                                (float*)d_out);
}
